// Head_11055245820529
// MI455X (gfx1250) — matrix operand order repaired and checked by reading, not compiled, not run on hardware
//
#include <hip/hip_runtime.h>

typedef __attribute__((ext_vector_type(16))) __bf16 v16bf;
typedef __attribute__((ext_vector_type(8)))  __bf16 v8bf;
typedef __attribute__((ext_vector_type(8)))  float  v8f;
typedef __attribute__((ext_vector_type(4)))  float  v4f;
typedef __attribute__((ext_vector_type(4)))  unsigned int u32x4;
typedef __attribute__((ext_vector_type(8)))  int i32x8;
typedef __attribute__((ext_vector_type(4)))  int i32x4;

namespace {
constexpr int kB   = 32;
constexpr int kT   = 1000;
constexpr int kD   = 1024;
constexpr int kH   = 64;
constexpr int kQT  = 63;   // 16-row tiles covering 1008 (padded) rows
constexpr int kKT  = 64;   // padded key-tile slots per batch (slot 63 never written)
constexpr int kS32 = 32;   // 32-key chunks per batch

// workspace layout, in bf16 elements (total ~12.3 MB)
constexpr size_t kWswzOff = 0;
constexpr size_t kWswzCnt = size_t(3) * 32 * 4 * 512;          // swizzled weights
constexpr size_t kQtOff   = kWswzOff + kWswzCnt;
constexpr size_t kQtCnt   = size_t(kB) * kQT * 2 * 512;        // Q^T, B-layout
constexpr size_t kKaOff   = kQtOff + kQtCnt;
constexpr size_t kKaCnt   = size_t(kB) * kKT * 2 * 512;        // K, A-layout
constexpr size_t kVbOff   = kKaOff + kKaCnt;                   // V, B-layout
} // namespace

// ---------------------------------------------------------------------------
// TDM: 1-D contiguous global->LDS DMA via Tensor Data Mover.
// D# group0: count=1, lds_addr, 57-bit global_addr, type=2 ("image").
// D# group1: data_size=8B, tensor_dim0=tile_dim0=units, tensor/tile_dim1=1.
// Tracked by TENSORcnt; in-order with other tensor ops from this wave.
// ---------------------------------------------------------------------------
__device__ __forceinline__ void tdm_load_1d(const __bf16* gsrc,
                                            unsigned lds_byte_off,
                                            unsigned bytes) {
  const unsigned units = bytes >> 3;                 // 8-byte elements
  const unsigned long long ga = (unsigned long long)(uintptr_t)gsrc;
  u32x4 g0;
  g0[0] = 1u;                                        // count=1 (valid user D#)
  g0[1] = lds_byte_off;                              // LDS destination (bytes)
  g0[2] = (unsigned)ga;                              // global_addr[31:0]
  g0[3] = (unsigned)((ga >> 32) & 0x1FFFFFFu) | (2u << 30);  // addr[56:32]|type=2
  i32x8 g1;
  g1[0] = 3 << 16;                                   // data_size = 8 bytes
  g1[1] = (int)((units & 0xFFFFu) << 16);            // tensor_dim0[15:0]
  g1[2] = (int)((units >> 16) | (1u << 16));         // tensor_dim0[31:16]|td1=1
  g1[3] = (int)((units & 0xFFFFu) << 16);            // tile_dim0
  g1[4] = 1;                                         // tile_dim1 = 1
  g1[5] = (int)units;                                // tensor_dim0_stride
  g1[6] = 0;
  g1[7] = 0;
  const i32x4 z4 = (i32x4){0, 0, 0, 0};
#if defined(__clang_major__) && (__clang_major__ >= 23)
  const i32x8 z8 = (i32x8){0, 0, 0, 0, 0, 0, 0, 0};
  __builtin_amdgcn_tensor_load_to_lds(g0, g1, z4, z4, z8, 0);
#else
  __builtin_amdgcn_tensor_load_to_lds(g0, g1, z4, z4, 0);
#endif
}

// ---------------------------------------------------------------------------
// Kernel 1: swizzle Wq/Wk/Wv [1024,64] fp32 -> bf16 B-layout tiles.
// ---------------------------------------------------------------------------
__global__ void prep_weights(const float* __restrict__ Wq,
                             const float* __restrict__ Wk,
                             const float* __restrict__ Wv,
                             __bf16* __restrict__ ws) {
  const int w = blockIdx.x * 8 + (threadIdx.x >> 5);   // 384 waves total
  const int l = threadIdx.x & 31;
  const int m  = w >> 7;
  const int kt = (w >> 2) & 31;
  const int j  = w & 3;
  const float* W = (m == 0) ? Wq : (m == 1) ? Wk : Wv;
  const int n  = j * 16 + (l & 15);
  const int kb = kt * 32 + ((l < 16) ? 0 : 8);
  __bf16* dst = ws + kWswzOff + (size_t)((m * 32 + kt) * 4 + j) * 512 + l * 16;
  v8bf lo, hi;
#pragma unroll
  for (int e = 0; e < 8; ++e) lo[e] = (__bf16)W[(size_t)(kb + e) * kH + n];
#pragma unroll
  for (int e = 0; e < 8; ++e) hi[e] = (__bf16)W[(size_t)(kb + 16 + e) * kH + n];
  *reinterpret_cast<v8bf*>(dst)     = lo;
  *reinterpret_cast<v8bf*>(dst + 8) = hi;
}

// ---------------------------------------------------------------------------
// Kernel 2: QKV projection. One wave per (batch, 16-row tile); computes all
// 12 output 16x16 tiles (3 matrices x 4 h-tiles) so x is read exactly once.
// All 12 B-tiles of an iteration are loaded before the WMMA burst so the
// scheduler can clause the loads and issue the 12 WMMAs back-to-back.
// ---------------------------------------------------------------------------
__global__ void qkv_project(const float* __restrict__ x, __bf16* __restrict__ ws) {
  __shared__ float smem[8 * 256];
  const int wslot = threadIdx.x >> 5;
  const int l     = threadIdx.x & 31;
  const int w     = blockIdx.x * 8 + wslot;            // 2016 waves
  const int it = w % kQT;
  const int b  = w / kQT;

  const int r  = it * 16 + (l & 15);
  const int rc = (r < kT) ? r : (kT - 1);              // clamp padded rows
  const float* row = x + ((size_t)b * kT + rc) * kD;
  const int c0 = (l < 16) ? 0 : 8;

  v8f acc[12];
#pragma unroll
  for (int mj = 0; mj < 12; ++mj) acc[mj] = (v8f){};

  for (int kk = 0; kk < 32; ++kk) {
    v16bf bv[12];
#pragma unroll
    for (int mj = 0; mj < 12; ++mj) {
      const int m = mj >> 2, j = mj & 3;
      bv[mj] = *reinterpret_cast<const v16bf*>(
          ws + kWswzOff + (size_t)(((m * 32 + kk) * 4) + j) * 512 + l * 16);
    }
    const float* p = row + kk * 32 + c0;
    v4f f0 = *reinterpret_cast<const v4f*>(p);
    v4f f1 = *reinterpret_cast<const v4f*>(p + 4);
    v4f f2 = *reinterpret_cast<const v4f*>(p + 16);
    v4f f3 = *reinterpret_cast<const v4f*>(p + 20);
    v16bf a;
#pragma unroll
    for (int e = 0; e < 4; ++e) {
      a[e]      = (__bf16)f0[e];
      a[4 + e]  = (__bf16)f1[e];
      a[8 + e]  = (__bf16)f2[e];
      a[12 + e] = (__bf16)f3[e];
    }
#pragma unroll
    for (int mj = 0; mj < 12; ++mj)
      acc[mj] = __builtin_amdgcn_wmma_f32_16x16x32_bf16(
          false, a, false, bv[mj], (short)0, acc[mj], false, false);
  }

  float* st = smem + wslot * 256;
#pragma unroll
  for (int mj = 0; mj < 12; ++mj) {
    const int m = mj >> 2, j = mj & 3;
    // stage D tile (C-layout) as plain [row m][col n] f32 in LDS
#pragma unroll
    for (int i = 0; i < 8; ++i)
      st[(i + ((l < 16) ? 0 : 8)) * 16 + (l & 15)] = acc[mj][i];
    // per-wave LDS pipeline is in-order; stop compiler reordering + wait
    asm volatile("s_wait_dscnt 0" ::: "memory");
    if (m == 0) {            // Q -> Q^T B-layout: tile (b, it, ht=j/2)
      __bf16* base = ws + kQtOff + (size_t)((b * kQT + it) * 2 + (j >> 1)) * 512;
      const int sec  = (j & 1) * 16;
      const int rowq = l >> 1;
      const int e0   = (l & 1) * 8;
      v8bf o;
#pragma unroll
      for (int u = 0; u < 8; ++u) o[u] = (__bf16)st[rowq * 16 + e0 + u];
      *reinterpret_cast<v8bf*>(base + (rowq + sec) * 16 + e0) = o;
    } else if (m == 1) {     // K -> A-layout: tile (b, it, ht=j/2)
      __bf16* base = ws + kKaOff + (size_t)((b * kKT + it) * 2 + (j >> 1)) * 512;
      const int eoff = (j & 1) * 8;
      const int srow = l & 15;
      const int hoff = (l < 16) ? 0 : 8;
      v8bf o;
#pragma unroll
      for (int u = 0; u < 8; ++u) o[u] = (__bf16)st[srow * 16 + hoff + u];
      *reinterpret_cast<v8bf*>(base + ((j & 1) * 16 + srow) * 16 + hoff) = o;
    } else {                 // V -> B-layout: tile (b, it/2, j); element-half by it&1
      __bf16* base = ws + kVbOff + (size_t)((b * kS32 + (it >> 1)) * 4 + j) * 512;
      const int lam = (l >> 1) + ((l & 1) ? 16 : 0);
      const int e0  = (l & 1) * 8;
      v8bf o;
#pragma unroll
      for (int u = 0; u < 8; ++u) o[u] = (__bf16)st[(e0 + u) * 16 + (l >> 1)];
      *reinterpret_cast<v8bf*>(base + lam * 16 + ((it & 1) ? 8 : 0)) = o;
    }
  }
}

// ---------------------------------------------------------------------------
// Kernel 3: flash attention, one wave per (batch, q-tile), 4 waves/block.
// K/V chunks (4KB+4KB, contiguous in the swizzled layouts) are double-buffered
// into LDS by the Tensor Data Mover: chunk t+1's two TDM ops are issued before
// chunk t is consumed, so the DMA overlaps the WMMA/softmax work.
// S^T = K(A) x Q^T(B): each query row lives in one lane (per-lane softmax),
// and the S^T C-layout regs map exactly onto the bf16 A-layout for P x V.
// ---------------------------------------------------------------------------
__global__ void attention(const __bf16* __restrict__ ws, float* __restrict__ out) {
  __shared__ __align__(16) __bf16 kv[4][2][4096];      // 4 waves x 2 bufs x 8KB
  const int l     = threadIdx.x & 31;
  const int wslot = threadIdx.x >> 5;
  const int w  = blockIdx.x * 4 + wslot;               // 2016 waves
  const int b  = w / kQT;
  const int it = w % kQT;

  const __bf16* Qt = ws + kQtOff + (size_t)((b * kQT + it) * 2) * 512;
  const v16bf qb0 = *reinterpret_cast<const v16bf*>(Qt + l * 16);
  const v16bf qb1 = *reinterpret_cast<const v16bf*>(Qt + 512 + l * 16);

  v8f o[4];
#pragma unroll
  for (int j = 0; j < 4; ++j) o[j] = (v8f){};
  float mx = -__builtin_inff();
  float lsum = 0.f;
  const int   qrow = it * 16 + (l & 15);
  const int   half = (l < 16) ? 0 : 8;
  const float c1   = 0.125f * 1.44269504088896f;       // scale * log2(e)
  const int   nch  = (it >> 1) + 1;                    // causal 32-key chunks

  const unsigned ldsBase = (unsigned)wslot * 2u * 8192u;  // kv[] starts at LDS 0
  // prologue: DMA chunk 0 into buffer 0
  tdm_load_1d(ws + kKaOff + (size_t)(b * kKT) * 2 * 512, ldsBase, 4096);
  tdm_load_1d(ws + kVbOff + (size_t)(b * kS32) * 4 * 512, ldsBase + 4096, 4096);

  for (int t = 0; t < nch; ++t) {
    const int buf = t & 1;
    if (t + 1 < nch) {       // DMA next chunk into the other buffer
      const unsigned nb = ldsBase + (unsigned)(buf ^ 1) * 8192u;
      tdm_load_1d(ws + kKaOff + (size_t)((b * kKT + 2 * (t + 1)) * 2) * 512, nb, 4096);
      tdm_load_1d(ws + kVbOff + (size_t)((b * kS32 + (t + 1)) * 4) * 512, nb + 4096, 4096);
      __builtin_amdgcn_s_wait_tensorcnt(2);  // current chunk's 2 ops retired
    } else {
      __builtin_amdgcn_s_wait_tensorcnt(0);
    }
    asm volatile("" ::: "memory");

    const __bf16* kb = &kv[wslot][buf][0];
    v16bf ka00 = *reinterpret_cast<const v16bf*>(kb + l * 16);
    v16bf ka01 = *reinterpret_cast<const v16bf*>(kb + 512 + l * 16);
    v16bf ka10 = *reinterpret_cast<const v16bf*>(kb + 1024 + l * 16);
    v16bf ka11 = *reinterpret_cast<const v16bf*>(kb + 1536 + l * 16);

    const v8f cz = (v8f){};
    v8f s0 = __builtin_amdgcn_wmma_f32_16x16x32_bf16(false, ka00, false, qb0,
                                                     (short)0, cz, false, false);
    s0 = __builtin_amdgcn_wmma_f32_16x16x32_bf16(false, ka01, false, qb1,
                                                 (short)0, s0, false, false);
    v8f s1 = __builtin_amdgcn_wmma_f32_16x16x32_bf16(false, ka10, false, qb0,
                                                     (short)0, cz, false, false);
    s1 = __builtin_amdgcn_wmma_f32_16x16x32_bf16(false, ka11, false, qb1,
                                                 (short)0, s1, false, false);

    // causal mask + scale, in log2 units; per-lane query row
    float tv[16];
    const int sbase = t * 32 + half;
#pragma unroll
    for (int i = 0; i < 8; ++i)
      tv[i] = ((sbase + i) <= qrow) ? s0[i] * c1 : -__builtin_inff();
#pragma unroll
    for (int i = 0; i < 8; ++i)
      tv[8 + i] = ((sbase + 16 + i) <= qrow) ? s1[i] * c1 : -__builtin_inff();

    float rmax = tv[0];
#pragma unroll
    for (int e = 1; e < 16; ++e) rmax = fmaxf(rmax, tv[e]);
    rmax = fmaxf(rmax, __shfl_xor(rmax, 16, 32));      // combine lane-halves
    const float nm   = fmaxf(mx, rmax);
    const float corr = __builtin_amdgcn_exp2f(mx - nm);

    float ps[16];
    float rsum = 0.f;
#pragma unroll
    for (int e = 0; e < 16; ++e) {
      ps[e] = __builtin_amdgcn_exp2f(tv[e] - nm);
      rsum += ps[e];
    }
    rsum += __shfl_xor(rsum, 16, 32);
    lsum = lsum * corr + rsum;
    mx = nm;

    // rescale O: broadcast per-row correction to C-layout rows (ds_bpermute)
    float cb[8];
#pragma unroll
    for (int i = 0; i < 8; ++i) cb[i] = __shfl(corr, i + half, 32);
#pragma unroll
    for (int j = 0; j < 4; ++j)
#pragma unroll
      for (int i = 0; i < 8; ++i) o[j][i] *= cb[i];

    // P: S^T C-layout regs are already in bf16 A-layout order
    v16bf pb;
#pragma unroll
    for (int e = 0; e < 16; ++e) pb[e] = (__bf16)ps[e];

    const __bf16* vbp = kb + 2048;
#pragma unroll
    for (int j = 0; j < 4; ++j) {
      v16bf vb = *reinterpret_cast<const v16bf*>(vbp + (size_t)j * 512 + l * 16);
      o[j] = __builtin_amdgcn_wmma_f32_16x16x32_bf16(false, pb, false, vb,
                                                     (short)0, o[j], false, false);
    }
  }

  // epilogue: normalize by row sums and store (f32 output)
  const float inv = 1.0f / lsum;
  float ib[8];
#pragma unroll
  for (int i = 0; i < 8; ++i) ib[i] = __shfl(inv, i + half, 32);
  const int hcol = l & 15;
#pragma unroll
  for (int j = 0; j < 4; ++j)
#pragma unroll
    for (int i = 0; i < 8; ++i) {
      const int q = it * 16 + i + half;
      if (q < kT)
        out[((size_t)b * kT + q) * kH + j * 16 + hcol] = o[j][i] * ib[i];
    }
}

// ---------------------------------------------------------------------------
extern "C" void kernel_launch(void* const* d_in, const int* in_sizes, int n_in,
                              void* d_out, int out_size, void* d_ws, size_t ws_size,
                              hipStream_t stream) {
  const float* x  = (const float*)d_in[0];
  const float* Wq = (const float*)d_in[1];
  const float* Wk = (const float*)d_in[2];
  const float* Wv = (const float*)d_in[3];
  float* out = (float*)d_out;
  __bf16* ws = (__bf16*)d_ws;

  prep_weights<<<dim3(48),  dim3(256), 0, stream>>>(Wq, Wk, Wv, ws); // 384 waves
  qkv_project <<<dim3(252), dim3(256), 0, stream>>>(x, ws);          // 2016 waves
  attention   <<<dim3(504), dim3(128), 0, stream>>>(ws, out);        // 2016 waves
}
